// VisionMamba_44238163149138
// MI455X (gfx1250) — compile-verified, matrix operand order checked by reading, not run on hardware
//
#include <hip/hip_runtime.h>

#define DEV __device__ __forceinline__

constexpr int cB = 8, cL = 1024, cDIM = 384, cDI = 768, cR = 24, cN = 16;
constexpr int cM   = cB * cL;     // 8192 token rows
constexpr int c2DI = 2 * cDI;     // 1536
constexpr int cXC  = cR + 2 * cN; // 56 proj cols per direction
constexpr float cEPS = 1e-5f;

typedef __attribute__((ext_vector_type(16))) _Float16 v16h;
typedef __attribute__((ext_vector_type(8)))  float    v8f;

DEV float siluf(float x) { return x / (1.f + __expf(-x)); }

// 16-half fragment per the CDNA5 16-bit A layout:
// elements 0..7  <- p[0..7]   (VGPR0..3, K = c0+0..7)
// elements 8..15 <- p[16..23] (VGPR4..7, K = c0+16..23)
DEV v16h load_frag16(const _Float16* p) {
  union { v16h v; uint4 q[2]; } u;
  u.q[0] = *(const uint4*)(p);
  u.q[1] = *(const uint4*)(p + 16);
  return u.v;
}

// One wave computes a 64x16 f32 block (4 M-tiles sharing each B fragment) of
// A(MxK) * B^T where B is (N x K) row-major.  Ktot is a compile-time constant
// at every call site: full unroll -> fresh SSA values per fragment (no register
// rotation movs, no WMMA->VALU hazard NOPs) and the scheduler pipelines the
// clause'd B128 loads ahead of the WMMAs.
DEV void wmma_tile_m4(const _Float16* __restrict__ A, int lda,
                      const _Float16* __restrict__ Bw, int ldb,
                      int m0, int n0, int Ktot, v8f acc[4]) {
  const int lane   = threadIdx.x & 31;
  const int half16 = lane >> 4;     // 0: K-chunk base 0, 1: base 8
  const int r      = lane & 15;
  const _Float16* ap = A  + (size_t)(m0 + r) * lda + half16 * 8;
  const _Float16* bp = Bw + (size_t)(n0 + r) * ldb + half16 * 8;
  v8f zero = {};
#pragma unroll
  for (int i = 0; i < 4; ++i) acc[i] = zero;
#pragma unroll
  for (int k = 0; k < Ktot; k += 32) {
    v16h bf = load_frag16(bp + k);
#pragma unroll
    for (int i = 0; i < 4; ++i) {
      v16h af = load_frag16(ap + (size_t)i * 16 * lda + k);
      acc[i] = __builtin_amdgcn_wmma_f32_16x16x32_f16(false, af, false, bf,
                                                      (short)0, acc[i], false, false);
    }
  }
}

// ---------------- conversion ----------------
__global__ __launch_bounds__(256)
void k_cvt(const float* __restrict__ src, _Float16* __restrict__ dst, int n) {
  int i = blockIdx.x * 256 + threadIdx.x;
  if (i < n) dst[i] = (_Float16)src[i];
}

// ---------------- GEMM 1: in_proj, fused split + SiLU(z) ----------------
__global__ __launch_bounds__(256)
void k_gemm_in(const _Float16* __restrict__ xh, const _Float16* __restrict__ w,
               _Float16* __restrict__ xraw, float* __restrict__ zs) {
  int tile = blockIdx.x * 8 + (threadIdx.x >> 5);
  const int tn_cnt = c2DI / 16;                 // 96
  int tm = tile / tn_cnt, tn = tile % tn_cnt;   // tm over 64-row blocks
  v8f acc[4];
  wmma_tile_m4(xh, cDIM, w, cDIM, tm * 64, tn * 16, cDIM, acc);
  int lane = threadIdx.x & 31;
  int n    = tn * 16 + (lane & 15);
  int mrow = tm * 64 + ((lane >> 4) << 3);
#pragma unroll
  for (int j = 0; j < 4; ++j)
#pragma unroll
    for (int i = 0; i < 8; ++i) {
      int m = mrow + j * 16 + i;
      float c = acc[j][i];
      if (n < cDI) xraw[(size_t)m * cDI + n] = (_Float16)c;
      else         zs[(size_t)m * cDI + (n - cDI)] = siluf(c);
    }
}

// ---------------- GEMM 2: 1x1 conv + bias + SiLU ----------------
__global__ __launch_bounds__(256)
void k_gemm_conv(const _Float16* __restrict__ xraw, const _Float16* __restrict__ w,
                 const float* __restrict__ bias,
                 float* __restrict__ xh32, _Float16* __restrict__ xh16) {
  int tile = blockIdx.x * 8 + (threadIdx.x >> 5);
  const int tn_cnt = cDI / 16;                  // 48
  int tm = tile / tn_cnt, tn = tile % tn_cnt;
  v8f acc[4];
  wmma_tile_m4(xraw, cDI, w, cDI, tm * 64, tn * 16, cDI, acc);
  int lane = threadIdx.x & 31;
  int n    = tn * 16 + (lane & 15);
  int mrow = tm * 64 + ((lane >> 4) << 3);
  float bn = bias[n];
#pragma unroll
  for (int j = 0; j < 4; ++j)
#pragma unroll
    for (int i = 0; i < 8; ++i) {
      int m = mrow + j * 16 + i;
      float s = siluf(acc[j][i] + bn);
      xh32[(size_t)m * cDI + n] = s;
      xh16[(size_t)m * cDI + n] = (_Float16)s;
    }
}

// ---------------- GEMM 3: x_proj (both directions: 112 cols) ----------------
__global__ __launch_bounds__(256)
void k_gemm_xproj(const _Float16* __restrict__ xh16, const _Float16* __restrict__ w,
                  float* __restrict__ xdbl) {
  int tile = blockIdx.x * 8 + (threadIdx.x >> 5);
  const int tn_cnt = (2 * cXC) / 16;            // 7
  int tm = tile / tn_cnt, tn = tile % tn_cnt;
  v8f acc[4];
  wmma_tile_m4(xh16, cDI, w, cDI, tm * 64, tn * 16, cDI, acc);
  int lane = threadIdx.x & 31;
  int n    = tn * 16 + (lane & 15);
  int mrow = tm * 64 + ((lane >> 4) << 3);
#pragma unroll
  for (int j = 0; j < 4; ++j)
#pragma unroll
    for (int i = 0; i < 8; ++i)
      xdbl[(size_t)(mrow + j * 16 + i) * (2 * cXC) + n] = acc[j][i];
}

// ---------------- selective scan ----------------
// grid: 48 blocks = (b: 8) x (k: 2) x (d-block: 3); thread <-> channel d,
// h[16] states in registers; dt_proj (R=24) fused as per-thread FMAs.
__global__ __launch_bounds__(256)
void k_scan(const float* __restrict__ xdbl, const float* __restrict__ xh32,
            const float* __restrict__ dtw_all, const float* __restrict__ dtb_all,
            const float* __restrict__ A_logs, float* __restrict__ yscan) {
  __shared__ float sdt[64][cR];
  __shared__ float sB[64][cN];
  __shared__ float sC[64][cN];
  int blk  = blockIdx.x;
  int dblk = blk % 3;
  int kdir = (blk / 3) & 1;
  int b    = blk / 6;
  int tid  = threadIdx.x;
  int d    = dblk * 256 + tid;

  float dtw[cR];
  const float* wp = dtw_all + ((size_t)kdir * cDI + d) * cR;
#pragma unroll
  for (int r = 0; r < cR; ++r) dtw[r] = wp[r];
  float bias = dtb_all[kdir * cDI + d];

  float Av[cN];
  const float* ap = A_logs + ((size_t)kdir * cDI + d) * cN;
#pragma unroll
  for (int n = 0; n < cN; ++n) Av[n] = -__expf(ap[n]);

  float h[cN];
#pragma unroll
  for (int n = 0; n < cN; ++n) h[n] = 0.f;

  const size_t plane = (size_t)cB * cL * cDI;
  for (int c0 = 0; c0 < cL; c0 += 64) {
    // prefetch next chunk's u-rows into cache while we stage this chunk
    if (c0 + 64 < cL) {
      int lnf   = c0 + 64 + (tid & 63);
      int rownf = kdir ? (cL - 1 - lnf) : lnf;
      __builtin_prefetch(&xh32[((size_t)b * cL + rownf) * cDI + d], 0, 3);
    }
    for (int idx = tid; idx < 64 * cXC; idx += 256) {
      int t = idx / cXC, c = idx % cXC;
      int l = c0 + t;
      int row = kdir ? (cL - 1 - l) : l;
      float v = xdbl[((size_t)b * cL + row) * (2 * cXC) + kdir * cXC + c];
      if (c < cR)            sdt[t][c] = v;
      else if (c < cR + cN)  sB[t][c - cR] = v;
      else                   sC[t][c - cR - cN] = v;
    }
    __syncthreads();
    for (int t = 0; t < 64; ++t) {
      int l = c0 + t;
      int row = kdir ? (cL - 1 - l) : l;
      float dt = bias;
#pragma unroll
      for (int r = 0; r < cR; ++r) dt = fmaf(sdt[t][r], dtw[r], dt);
      float delta = (dt > 20.f) ? dt : log1pf(__expf(dt));   // softplus
      float u  = xh32[((size_t)b * cL + row) * cDI + d];
      float du = delta * u;
      float y  = 0.f;
#pragma unroll
      for (int n = 0; n < cN; ++n) {
        float dA = __expf(delta * Av[n]);
        h[n] = fmaf(dA, h[n], du * sB[t][n]);
        y    = fmaf(h[n], sC[t][n], y);
      }
      // store at ORIGINAL position -> merge is a plain add later
      yscan[(size_t)kdir * plane + ((size_t)b * cL + row) * cDI + d] = y;
    }
    __syncthreads();
  }
}

// ---------------- merge + LayerNorm + z-gate ----------------
__global__ __launch_bounds__(256)
void k_ln_gate(const float* __restrict__ yscan, const float* __restrict__ xh32,
               const float* __restrict__ zs, const float* __restrict__ Ds,
               const float* __restrict__ nw, const float* __restrict__ nb,
               _Float16* __restrict__ yh) {
  __shared__ float ss[256], sq[256];
  int row = blockIdx.x, tid = threadIdx.x;
  const size_t plane = (size_t)cB * cL * cDI;
  float v[3];
  float s = 0.f, q = 0.f;
#pragma unroll
  for (int i = 0; i < 3; ++i) {
    int d = tid + i * 256;
    size_t idx = (size_t)row * cDI + d;
    float val = yscan[idx] + yscan[plane + idx] + (Ds[d] + Ds[cDI + d]) * xh32[idx];
    v[i] = val; s += val; q += val * val;
  }
  ss[tid] = s; sq[tid] = q;
  __syncthreads();
  for (int off = 128; off > 0; off >>= 1) {
    if (tid < off) { ss[tid] += ss[tid + off]; sq[tid] += sq[tid + off]; }
    __syncthreads();
  }
  float mu  = ss[0] * (1.f / cDI);
  float var = sq[0] * (1.f / cDI) - mu * mu;
  float rs  = rsqrtf(var + cEPS);
#pragma unroll
  for (int i = 0; i < 3; ++i) {
    int d = tid + i * 256;
    size_t idx = (size_t)row * cDI + d;
    float o = (v[i] - mu) * rs * nw[d] + nb[d];
    o *= zs[idx];
    yh[idx] = (_Float16)o;
  }
}

// ---------------- GEMM 4: out_proj ----------------
__global__ __launch_bounds__(256)
void k_gemm_out(const _Float16* __restrict__ yh, const _Float16* __restrict__ w,
                float* __restrict__ out) {
  int tile = blockIdx.x * 8 + (threadIdx.x >> 5);
  const int tn_cnt = cDIM / 16;                 // 24
  int tm = tile / tn_cnt, tn = tile % tn_cnt;
  v8f acc[4];
  wmma_tile_m4(yh, cDI, w, cDI, tm * 64, tn * 16, cDI, acc);
  int lane = threadIdx.x & 31;
  int n    = tn * 16 + (lane & 15);
  int mrow = tm * 64 + ((lane >> 4) << 3);
#pragma unroll
  for (int j = 0; j < 4; ++j)
#pragma unroll
    for (int i = 0; i < 8; ++i)
      out[(size_t)(mrow + j * 16 + i) * cDIM + n] = acc[j][i];
}

extern "C" void kernel_launch(void* const* d_in, const int* in_sizes, int n_in,
                              void* d_out, int out_size, void* d_ws, size_t ws_size,
                              hipStream_t stream) {
  const float* x       = (const float*)d_in[0];   // (B,L,DIM)
  const float* w_in    = (const float*)d_in[1];   // (2DI,DIM)
  const float* w_conv  = (const float*)d_in[2];   // (DI,DI)
  const float* b_conv  = (const float*)d_in[3];   // (DI)
  const float* w_xproj = (const float*)d_in[4];   // (K,56,DI) == (112,DI)
  const float* w_dt    = (const float*)d_in[5];   // (K,DI,R)
  const float* b_dt    = (const float*)d_in[6];   // (K,DI)
  const float* A_logs  = (const float*)d_in[7];   // (K*DI,N)
  const float* Ds      = (const float*)d_in[8];   // (K*DI)
  const float* nw      = (const float*)d_in[9];   // (DI)
  const float* nb      = (const float*)d_in[10];  // (DI)
  const float* w_out   = (const float*)d_in[11];  // (DIM,DI)
  float* out = (float*)d_out;

  char* ws = (char*)d_ws;
  size_t off = 0;
  auto take = [&](size_t bytes) -> void* {
    void* p = ws + off;
    off = (off + bytes + 255) & ~(size_t)255;
    return p;
  };

  _Float16* x_h      = (_Float16*)take((size_t)cM * cDIM * 2);
  _Float16* w_in_h   = (_Float16*)take((size_t)c2DI * cDIM * 2);
  _Float16* w_conv_h = (_Float16*)take((size_t)cDI * cDI * 2);
  _Float16* w_xp_h   = (_Float16*)take((size_t)2 * cXC * cDI * 2);
  _Float16* w_out_h  = (_Float16*)take((size_t)cDIM * cDI * 2);
  _Float16* xraw_h   = (_Float16*)take((size_t)cM * cDI * 2);
  float*    zs       = (float*)take((size_t)cM * cDI * 4);
  float*    xh32     = (float*)take((size_t)cM * cDI * 4);
  _Float16* xh16     = (_Float16*)take((size_t)cM * cDI * 2);
  float*    xdbl     = (float*)take((size_t)cM * 2 * cXC * 4);
  float*    yscan    = (float*)take((size_t)2 * cM * cDI * 4);
  _Float16* y_h      = (_Float16*)take((size_t)cM * cDI * 2);

  auto cvt = [&](const float* s, _Float16* d, int n) {
    k_cvt<<<(n + 255) / 256, 256, 0, stream>>>(s, d, n);
  };
  cvt(x,       x_h,      cM * cDIM);
  cvt(w_in,    w_in_h,   c2DI * cDIM);
  cvt(w_conv,  w_conv_h, cDI * cDI);
  cvt(w_xproj, w_xp_h,   2 * cXC * cDI);
  cvt(w_out,   w_out_h,  cDIM * cDI);

  // in_proj: (8192x384)@(384x1536): (8192/64)*96 = 12288 wave-jobs / 8 waves
  k_gemm_in<<<1536, 256, 0, stream>>>(x_h, w_in_h, xraw_h, zs);
  // conv:    (8192x768)@(768x768): 128*48 = 6144 wave-jobs
  k_gemm_conv<<<768, 256, 0, stream>>>(xraw_h, w_conv_h, b_conv, xh32, xh16);
  // x_proj:  (8192x768)@(768x112): 128*7 = 896 wave-jobs
  k_gemm_xproj<<<112, 256, 0, stream>>>(xh16, w_xp_h, xdbl);
  // selective scan: 8(b) x 2(k) x 3(d-block)
  k_scan<<<48, 256, 0, stream>>>(xdbl, xh32, w_dt, b_dt, A_logs, yscan);
  // merge + LN + gate
  k_ln_gate<<<cM, 256, 0, stream>>>(yscan, xh32, zs, Ds, nw, nb, y_h);
  // out_proj: (8192x768)@(768x384): 128*24 = 3072 wave-jobs
  k_gemm_out<<<384, 256, 0, stream>>>(y_h, w_out_h, out);
  (void)in_sizes; (void)n_in; (void)out_size; (void)ws_size;
}
